// CausalSelfAttention_10857677324324
// MI455X (gfx1250) — compile-verified, matrix operand order checked by reading, not run on hardware
//
#include <hip/hip_runtime.h>
#include <math.h>

// ---------------------------------------------------------------------------
// CDNA5 (gfx1250) GQA sliding-window attention, all matmuls via
// v_wmma_f32_16x16x32_f16 (wave32).
//
// Shapes (fixed by reference): B=2, T=2048, C=1024, H=16, d=64, KV=256,
// WINDOW=256.
//
// Round-2 change: GEMM wave blocking 2x2 -> 2x4 (32x64 per wave, 128x128 per
// 8-wave block): 12 global_load_b128 per 8 WMMAs instead of 16 per 8, to keep
// the XDL pipe fed from the (fully L2-resident) operands.
// ---------------------------------------------------------------------------

typedef __attribute__((ext_vector_type(16))) _Float16 v16h;
typedef __attribute__((ext_vector_type(8)))  float    v8f;

union V8 { v8f v; float f[8]; };

#define WMMA_F16(a, b, c) \
  __builtin_amdgcn_wmma_f32_16x16x32_f16(false, (a), false, (b), (short)0, (c), false, false)

// Load a 16-half A/B fragment: two contiguous 16-byte runs
// (halves 0..7 from p0, halves 8..15 from p1) — matches the 16-bit operand
// VGPR layout (K = kb..kb+7 then 16+kb..16+kb+7 for this lane).
__device__ __forceinline__ v16h load_frag(const _Float16* p0, const _Float16* p1) {
  union { uint4 q[2]; v16h v; } u;
  u.q[0] = *(const uint4*)p0;
  u.q[1] = *(const uint4*)p1;
  return u.v;
}

__device__ __forceinline__ int imax(int a, int b) { return a > b ? a : b; }
__device__ __forceinline__ int imin(int a, int b) { return a < b ? a : b; }

// ---------------------------------------------------------------------------
// Elementwise f32 -> f16
// ---------------------------------------------------------------------------
__global__ void f32_to_f16_kernel(const float* __restrict__ src,
                                  _Float16* __restrict__ dst, int n) {
  int i = blockIdx.x * 256 + threadIdx.x;
  if (i < n) dst[i] = (_Float16)src[i];
}

// ---------------------------------------------------------------------------
// Transpose-convert: src f32 [K][N] row-major  ->  dst f16 [N][K] row-major
// ---------------------------------------------------------------------------
__global__ void transpose_f16_kernel(const float* __restrict__ src,
                                     _Float16* __restrict__ dst, int K, int N) {
  int idx = blockIdx.x * 256 + threadIdx.x;
  if (idx < K * N) {
    int k = idx % K;
    int n = idx / K;
    dst[idx] = (_Float16)src[(size_t)k * N + n];
  }
}

// ---------------------------------------------------------------------------
// f16 WMMA GEMM:  out = A[MxK] * B[KxN] + bias,  B given transposed (Bt[NxK]).
// Block = 256 threads = 8 waves (4 in M x 2 in N); each wave computes a
// 32x64 tile (2x4 WMMA tiles), block covers 128x128.
// mode: 0 = f16 row-major out, 1 = f16 out transposed per batch
// ([b][n][t], t = row % Tdim), 2 = f32 row-major out.
// ---------------------------------------------------------------------------
__global__ __launch_bounds__(256) void gemm_f16_kernel(
    const _Float16* __restrict__ A, const _Float16* __restrict__ Bt,
    const float* __restrict__ bias, _Float16* __restrict__ outH,
    float* __restrict__ outF, int M, int N, int K, int mode, int Tdim) {
  const int lane = threadIdx.x & 31;
  const int wave = threadIdx.x >> 5;
  const int wm = wave & 3, wn = wave >> 2;
  const int row0 = blockIdx.x * 128 + wm * 32;
  const int col0 = blockIdx.y * 128 + wn * 64;
  const int lr = lane & 15;
  const int hg = lane >> 4;   // half-wave group
  const int kb = hg * 8;      // per-lane K sub-offset

  V8 c[2][4];
#pragma unroll
  for (int mi = 0; mi < 2; ++mi)
#pragma unroll
    for (int ni = 0; ni < 4; ++ni)
#pragma unroll
      for (int i = 0; i < 8; ++i) c[mi][ni].f[i] = 0.0f;

  const _Float16* arow[2];
  arow[0] = A + (size_t)(row0 + lr) * K + kb;
  arow[1] = A + (size_t)(row0 + 16 + lr) * K + kb;
  const _Float16* bcol[4];
#pragma unroll
  for (int ni = 0; ni < 4; ++ni)
    bcol[ni] = Bt + (size_t)(col0 + ni * 16 + lr) * K + kb;

  for (int kc = 0; kc < K; kc += 32) {
    v16h a[2], bf[4];
#pragma unroll
    for (int mi = 0; mi < 2; ++mi)
      a[mi] = load_frag(arow[mi] + kc, arow[mi] + kc + 16);
#pragma unroll
    for (int ni = 0; ni < 4; ++ni)
      bf[ni] = load_frag(bcol[ni] + kc, bcol[ni] + kc + 16);
#pragma unroll
    for (int mi = 0; mi < 2; ++mi)
#pragma unroll
      for (int ni = 0; ni < 4; ++ni)
        c[mi][ni].v = WMMA_F16(a[mi], bf[ni], c[mi][ni].v);
  }

  float bv[4];
#pragma unroll
  for (int ni = 0; ni < 4; ++ni) bv[ni] = bias[col0 + ni * 16 + lr];

#pragma unroll
  for (int mi = 0; mi < 2; ++mi) {
#pragma unroll
    for (int i = 0; i < 8; ++i) {
      const int r = row0 + mi * 16 + i + 8 * hg;
#pragma unroll
      for (int ni = 0; ni < 4; ++ni) {
        const int cc = col0 + ni * 16 + lr;
        const float v = c[mi][ni].f[i] + bv[ni];
        if (mode == 0) {
          outH[(size_t)r * N + cc] = (_Float16)v;
        } else if (mode == 1) {
          const int bi = r / Tdim, t = r % Tdim;
          outH[((size_t)bi * N + cc) * Tdim + t] = (_Float16)v;
        } else {
          outF[(size_t)r * N + cc] = v;
        }
      }
    }
  }
}

// ---------------------------------------------------------------------------
// Banded flash attention.  One wave per 16-row q-tile.  d=64, WINDOW=256.
// 9 uniform chunks of 32 keys cover [qi0-272, qi0+15] (uniform -> no
// divergence).  qh [B*T][1024], kh [B*T][256], vt [B*256][2048] (V transposed),
// yh [B*T][1024].
// ---------------------------------------------------------------------------
__global__ __launch_bounds__(256) void attn_kernel(
    const _Float16* __restrict__ qh, const _Float16* __restrict__ kh,
    const _Float16* __restrict__ vt, _Float16* __restrict__ yh) {
  __shared__ _Float16 pbuf[8][16 * 32];   // per-wave P staging (1KB each)

  const int lane = threadIdx.x & 31;
  const int wave = threadIdx.x >> 5;
  const int wid = blockIdx.x * 8 + wave;  // 4096 waves total
  const int qt = wid & 127;               // T/16 = 128 q-tiles
  const int h  = (wid >> 7) & 15;
  const int b  = wid >> 11;
  const int g  = h >> 2;                  // KV group (repeat_interleave)
  const int qi0 = qt * 16;
  const int lr = lane & 15;
  const int hg = lane >> 4;
  const int kb = hg * 8;

  // Q A-fragments for d-chunks [0,32) and [32,64); rows qi0+lr.
  const _Float16* qbase =
      qh + ((size_t)(b * 2048) + qi0 + lr) * 1024 + h * 64;
  const v16h qa0 = load_frag(qbase + kb, qbase + kb + 16);
  const v16h qa1 = load_frag(qbase + 32 + kb, qbase + 32 + kb + 16);

  V8 o0 = {}, o1 = {}, o2 = {}, o3 = {};
  float m[8], l[8];
#pragma unroll
  for (int i = 0; i < 8; ++i) { m[i] = -INFINITY; l[i] = 0.0f; }

  _Float16* pw = &pbuf[wave][0];

  for (int chunk = 0; chunk < 9; ++chunk) {
    const int kbase = qi0 - 272 + chunk * 32;

    // ---- S = Q K^T for 2 key tiles of 16 (K-dim = d = 64 -> 2 WMMA each)
    const int j0 = kbase + lr;            // key pos for score column (tile 0)
    const int j1 = j0 + 16;               // tile 1
    const int j0c = imin(imax(j0, 0), 2047);
    const int j1c = imin(imax(j1, 0), 2047);
    const _Float16* kp0 = kh + ((size_t)(b * 2048) + j0c) * 256 + g * 64;
    const _Float16* kp1 = kh + ((size_t)(b * 2048) + j1c) * 256 + g * 64;
    v16h kf00 = load_frag(kp0 + kb, kp0 + kb + 16);
    v16h kf01 = load_frag(kp0 + 32 + kb, kp0 + 32 + kb + 16);
    v16h kf10 = load_frag(kp1 + kb, kp1 + kb + 16);
    v16h kf11 = load_frag(kp1 + 32 + kb, kp1 + 32 + kb + 16);

    V8 s0 = {}, s1 = {};
    s0.v = WMMA_F16(qa0, kf00, s0.v);
    s0.v = WMMA_F16(qa1, kf01, s0.v);
    s1.v = WMMA_F16(qa0, kf10, s1.v);
    s1.v = WMMA_F16(qa1, kf11, s1.v);

    // ---- online softmax (row reductions across the 16 lanes of each half)
#pragma unroll
    for (int i = 0; i < 8; ++i) {
      const int rowi = qi0 + i + 8 * hg;
      float sv0 = s0.f[i] * 0.125f;       // 1/sqrt(64)
      float sv1 = s1.f[i] * 0.125f;
      const bool v0 = (j0 >= 0) && (j0 <= rowi) && (j0 >= rowi - 256);
      const bool v1 = (j1 >= 0) && (j1 <= rowi) && (j1 >= rowi - 256);
      if (!v0) sv0 = -INFINITY;
      if (!v1) sv1 = -INFINITY;
      float rmax = fmaxf(sv0, sv1);
#pragma unroll
      for (int x = 1; x < 16; x <<= 1) rmax = fmaxf(rmax, __shfl_xor(rmax, x, 32));
      const float mn = fmaxf(m[i], rmax);
      const float p0 = v0 ? expf(sv0 - mn) : 0.0f;
      const float p1 = v1 ? expf(sv1 - mn) : 0.0f;
      float rsum = p0 + p1;
#pragma unroll
      for (int x = 1; x < 16; x <<= 1) rsum += __shfl_xor(rsum, x, 32);
      const float sf = (m[i] == mn) ? 1.0f : expf(m[i] - mn);
      l[i] = l[i] * sf + rsum;
      m[i] = mn;
      o0.f[i] *= sf; o1.f[i] *= sf; o2.f[i] *= sf; o3.f[i] *= sf;
      const int prow = i + 8 * hg;
      pw[prow * 32 + lr]      = (_Float16)p0;
      pw[prow * 32 + 16 + lr] = (_Float16)p1;
    }

    // wave-private LDS: wait for DS stores, then reload as A-fragment
    asm volatile("s_wait_dscnt 0" ::: "memory");
    const _Float16* pp = pw + lr * 32 + kb;
    const v16h pa = load_frag(pp, pp + 16);

    // ---- O += P V ; V B-frags from transposed vt: contiguous in key pos.
    // 8-aligned runs => a run is either fully valid or fully masked (P=0),
    // so clamping the start to 0 is safe.
    const int jb0 = imax(kbase + kb, 0);
    const int jb1 = imax(kbase + kb + 16, 0);
    const _Float16* vbase = vt + ((size_t)(b * 256) + g * 64 + lr) * 2048;
    v16h vf0 = load_frag(vbase + jb0, vbase + jb1);
    v16h vf1 = load_frag(vbase + 16 * 2048 + jb0, vbase + 16 * 2048 + jb1);
    v16h vf2 = load_frag(vbase + 32 * 2048 + jb0, vbase + 32 * 2048 + jb1);
    v16h vf3 = load_frag(vbase + 48 * 2048 + jb0, vbase + 48 * 2048 + jb1);
    o0.v = WMMA_F16(pa, vf0, o0.v);
    o1.v = WMMA_F16(pa, vf1, o1.v);
    o2.v = WMMA_F16(pa, vf2, o2.v);
    o3.v = WMMA_F16(pa, vf3, o3.v);
  }

  // ---- normalize and store (f16, row-major [B*T][1024])
  _Float16* yb = yh + ((size_t)(b * 2048) + qi0) * 1024 + h * 64;
#pragma unroll
  for (int i = 0; i < 8; ++i) {
    const float inv = 1.0f / l[i];
    const size_t r = (size_t)(i + 8 * hg) * 1024;
    yb[r + lr]      = (_Float16)(o0.f[i] * inv);
    yb[r + 16 + lr] = (_Float16)(o1.f[i] * inv);
    yb[r + 32 + lr] = (_Float16)(o2.f[i] * inv);
    yb[r + 48 + lr] = (_Float16)(o3.f[i] * inv);
  }
}

// ---------------------------------------------------------------------------
// Host-side orchestration
// ---------------------------------------------------------------------------
extern "C" void kernel_launch(void* const* d_in, const int* in_sizes, int n_in,
                              void* d_out, int out_size, void* d_ws, size_t ws_size,
                              hipStream_t stream) {
  (void)in_sizes; (void)n_in; (void)out_size; (void)ws_size;
  const float* x  = (const float*)d_in[0];
  const float* Wq = (const float*)d_in[1];
  const float* bq = (const float*)d_in[2];
  const float* Wk = (const float*)d_in[3];
  const float* bk = (const float*)d_in[4];
  const float* Wv = (const float*)d_in[5];
  const float* bv = (const float*)d_in[6];
  const float* Wo = (const float*)d_in[7];
  const float* bo = (const float*)d_in[8];
  float* out = (float*)d_out;

  const int B = 2, T = 2048, C = 1024, KV = 256;
  const int MT = B * T;                       // 4096 rows

  // f16 workspace layout
  _Float16* xh  = (_Float16*)d_ws;
  _Float16* Wqt = xh  + (size_t)MT * C;       // x as f16            (4M halves)
  _Float16* Wkt = Wqt + (size_t)C * C;        // Wq^T f16            (1M)
  _Float16* Wvt = Wkt + (size_t)KV * C;       // Wk^T f16            (256K)
  _Float16* Wot = Wvt + (size_t)KV * C;       // Wv^T f16            (256K)
  _Float16* qh  = Wot + (size_t)C * C;        // Wo^T f16            (1M)
  _Float16* khp = qh  + (size_t)MT * C;       // Q f16               (4M)
  _Float16* vtp = khp + (size_t)MT * KV;      // K f16               (1M)
  _Float16* yh  = vtp + (size_t)B * KV * T;   // V^T f16 [B][KV][T]  (1M)
                                              // yh: attn out f16    (4M)

  // 1) conversions / transposes
  f32_to_f16_kernel<<<(MT * C) / 256, 256, 0, stream>>>(x, xh, MT * C);
  transpose_f16_kernel<<<(C * C) / 256, 256, 0, stream>>>(Wq, Wqt, C, C);
  transpose_f16_kernel<<<(C * KV) / 256, 256, 0, stream>>>(Wk, Wkt, C, KV);
  transpose_f16_kernel<<<(C * KV) / 256, 256, 0, stream>>>(Wv, Wvt, C, KV);
  transpose_f16_kernel<<<(C * C) / 256, 256, 0, stream>>>(Wo, Wot, C, C);

  // 2) QKV projections (WMMA GEMMs), block tile 128x128
  gemm_f16_kernel<<<dim3(MT / 128, C / 128), 256, 0, stream>>>(
      xh, Wqt, bq, qh, nullptr, MT, C, C, /*mode=*/0, T);
  gemm_f16_kernel<<<dim3(MT / 128, KV / 128), 256, 0, stream>>>(
      xh, Wkt, bk, khp, nullptr, MT, KV, C, /*mode=*/0, T);
  gemm_f16_kernel<<<dim3(MT / 128, KV / 128), 256, 0, stream>>>(
      xh, Wvt, bv, vtp, nullptr, MT, KV, C, /*mode=*/1, T);   // V transposed

  // 3) banded flash attention: 4096 q-tile waves / 8 waves per block
  attn_kernel<<<512, 256, 0, stream>>>(qh, khp, vtp, yh);

  // 4) output projection, f32 + bias straight into d_out
  gemm_f16_kernel<<<dim3(MT / 128, C / 128), 256, 0, stream>>>(
      yh, Wot, bo, nullptr, out, MT, C, C, /*mode=*/2, T);
}
